// Edge3DEmbedding_6949257085354
// MI455X (gfx1250) — matrix operand order repaired and checked by reading, not compiled, not run on hardware
//
#include <hip/hip_runtime.h>

#define B_ 2
#define N_ 512
#define K_ 128
#define E_ 768

typedef __attribute__((ext_vector_type(16))) _Float16 v16h;
typedef __attribute__((ext_vector_type(8)))  float    v8f;

// ---------------------------------------------------------------------------
// Kernel 1: time-step encoder. grid=(B,2): y==0 -> t=time_pos[b], y==1 -> t=0.
// Writes 2*t_enc(t) into tbuf[(variant*B + b)*K + k].
// ---------------------------------------------------------------------------
__global__ __launch_bounds__(128) void time_enc_kernel(
    const int* __restrict__ time_pos,
    const float* __restrict__ w1, const float* __restrict__ b1,
    const float* __restrict__ w2, const float* __restrict__ b2,
    float* __restrict__ tbuf)
{
    __shared__ float e[K_];
    __shared__ float h[K_];
    const int b = blockIdx.x;
    const int variant = blockIdx.y;
    const int k = threadIdx.x;
    const float t = (variant == 0) ? (float)time_pos[b] : 0.0f;
    const float log10000 = 9.210340371976184f;

    if (k < 64) {
        float freq = __expf(-log10000 * (float)k * (1.0f / 64.0f));
        e[k] = __cosf(t * freq);
    } else {
        float freq = __expf(-log10000 * (float)(k - 64) * (1.0f / 64.0f));
        e[k] = __sinf(t * freq);
    }
    __syncthreads();

    float acc = b1[k];
    #pragma unroll 4
    for (int m = 0; m < K_; ++m) acc += e[m] * w1[m * K_ + k];
    h[k] = acc / (1.0f + __expf(-acc));          // SiLU
    __syncthreads();

    float acc2 = b2[k];
    #pragma unroll 4
    for (int m = 0; m < K_; ++m) acc2 += h[m] * w2[m * K_ + k];
    tbuf[(variant * B_ + b) * K_ + k] = 2.0f * acc2;
}

// ---------------------------------------------------------------------------
// Kernel 2: fused delta_pos + Gaussian edge features + j-reduction.
// One block per (b,i); 8 waves; wave w handles j = jb + w; lane owns 4 k's.
// ---------------------------------------------------------------------------
__global__ __launch_bounds__(256) void edge_kernel(
    const float* __restrict__ pos,              // [B,N,3]
    const int*   __restrict__ nte,              // [B,N,N,2]
    const unsigned char* __restrict__ padding_mask, // [B,N] bool
    const unsigned char* __restrict__ mask_aa,      // [B,N,1] bool
    const unsigned char* __restrict__ mask_pos,     // [B,N,1] bool
    const float* __restrict__ means,            // [K]
    const float* __restrict__ stds,             // [K]
    const float* __restrict__ mul_w,            // [NUM_EDGES]
    const float* __restrict__ bias_w,           // [NUM_EDGES]
    const float* __restrict__ tbuf,             // [2,B,K] (already 2*t)
    float* __restrict__ edge_out,               // [B,N,N,K]
    float* __restrict__ delta_out,              // [B,N,N,3]
    _Float16* __restrict__ sum_out)             // [B*N, K] f16 staging
{
    __shared__ float red[8][K_];

    const int bi   = blockIdx.x;
    const int b    = bi / N_;
    const int i    = bi % N_;
    const int tid  = threadIdx.x;
    const int lane = tid & 31;
    const int wave = tid >> 5;
    const int k0   = lane * 4;

    // per-lane k constants (mean, 1/std, 1/(sqrt(2*pi)*std), time vectors)
    const float A_CONST = 2.50662682463f;   // sqrt(2 * 3.14159)
    float m4[4], is4[4], cf4[4], tt[4], tt0[4];
    #pragma unroll
    for (int c = 0; c < 4; ++c) {
        float s  = fabsf(stds[k0 + c]) + 0.01f;
        m4[c]  = means[k0 + c];
        is4[c] = 1.0f / s;
        cf4[c] = 1.0f / (A_CONST * s);
        tt[c]  = tbuf[b * K_ + k0 + c];           // 2*t_enc(t)
        tt0[c] = tbuf[(B_ + b) * K_ + k0 + c];    // 2*t_enc(0)
    }

    const float px = pos[(b * N_ + i) * 3 + 0];
    const float py = pos[(b * N_ + i) * 3 + 1];
    const float pz = pos[(b * N_ + i) * 3 + 2];
    const bool maa_i  = mask_aa[b * N_ + i] != 0;
    const bool mpos_i = mask_pos[b * N_ + i] != 0;

    float s4[4] = {0.0f, 0.0f, 0.0f, 0.0f};

    for (int jb = 0; jb < N_; jb += 8) {
        const int j = jb + wave;
        const float qx = pos[(b * N_ + j) * 3 + 0];
        const float qy = pos[(b * N_ + j) * 3 + 1];
        const float qz = pos[(b * N_ + j) * 3 + 2];
        const float dx = qx - px, dy = qy - py, dz = qz - pz;
        const float dist = sqrtf(dx * dx + dy * dy + dz * dz);
        const float inv  = 1.0f / (dist + 1e-5f);

        const size_t base = (size_t)(b * N_ + i) * N_ + j;

        if (lane < 3) {
            float comp = (lane == 0) ? dx : ((lane == 1) ? dy : dz);
            delta_out[base * 3 + lane] = comp * inv;
        }

        int e0 = nte[base * 2 + 0];
        int e1 = nte[base * 2 + 1];
        const bool maa_j = mask_aa[b * N_ + j] != 0;
        if (maa_i) e0 = 0;
        if (maa_j) e1 = 0;
        const float mul = mul_w[e0] + mul_w[e1];
        const float bia = bias_w[e0] + bias_w[e1];
        const float x   = mul * dist + bia;

        const bool pad_j = padding_mask[b * N_ + j] != 0;
        const bool md    = mpos_i || (mask_pos[b * N_ + j] != 0);

        float4 out;
        float o[4];
        #pragma unroll
        for (int c = 0; c < 4; ++c) {
            float d = (x - m4[c]) * is4[c];
            float f = cf4[c] * __expf(-0.5f * d * d) + (md ? tt[c] : tt0[c]);
            if (pad_j) f = 0.0f;
            o[c] = f;
            s4[c] += f;
        }
        out.x = o[0]; out.y = o[1]; out.z = o[2]; out.w = o[3];
        *(float4*)(edge_out + base * K_ + k0) = out;   // 16B/lane, b128 store
    }

    // cross-wave reduction of the j-sums
    #pragma unroll
    for (int c = 0; c < 4; ++c) red[wave][k0 + c] = s4[c];
    __syncthreads();
    if (tid < K_) {
        float s = 0.0f;
        #pragma unroll
        for (int w = 0; w < 8; ++w) s += red[w][tid];
        sum_out[(size_t)bi * K_ + tid] = (_Float16)s;
    }
}

// ---------------------------------------------------------------------------
// Kernel 3: merge = sum_edge [1024x128] @ proj_w [128x768] + proj_b, masked.
// One wave (32 threads) per 16x16 tile; 4 x v_wmma_f32_16x16x32_f16 over K.
// Fragment layouts per CDNA5 ISA 7.12.2.
// ---------------------------------------------------------------------------
__global__ __launch_bounds__(32) void merge_wmma_kernel(
    const _Float16* __restrict__ Af,            // [1024,128] f16
    const float* __restrict__ proj_w,           // [128,768]
    const float* __restrict__ proj_b,           // [768]
    const unsigned char* __restrict__ padding_mask, // [1024]
    float* __restrict__ merge)                  // [1024,768]
{
    const int tileN = blockIdx.x;   // 0..47
    const int tileM = blockIdx.y;   // 0..63
    const int l  = threadIdx.x;
    const int nl = l & 15;
    const int hi = l >> 4;
    const int row0 = tileM * 16;
    const int col0 = tileN * 16;

    v8f c = {};
    #pragma unroll
    for (int kk = 0; kk < 4; ++kk) {
        const int kbase = kk * 32;
        v16h a, bb;
        // A fragment: lane's row = row0 + (l&15); VGPR p holds K pairs
        // p<4 -> K = 2p + 8*hi + {0,1};  p>=4 -> K = 16 + 2(p-4) + 8*hi + {0,1}
        #pragma unroll
        for (int p = 0; p < 8; ++p) {
            const int kgrp = (p < 4) ? (2 * p) : (16 + 2 * (p - 4));
            #pragma unroll
            for (int s = 0; s < 2; ++s) {
                const int koff = kgrp + hi * 8 + s;
                a[2 * p + s] = Af[(size_t)(row0 + nl) * K_ + kbase + koff];
            }
        }
        // B fragment: lane's col = col0 + (l&15); hh<8 -> K = 8*hi + hh; hh>=8 -> K = 16 + 8*hi + (hh-8)
        #pragma unroll
        for (int hh = 0; hh < 16; ++hh) {
            const int kr = kbase + ((hh < 8) ? (hi * 8 + hh) : (16 + hi * 8 + (hh - 8)));
            bb[hh] = (_Float16)proj_w[(size_t)kr * E_ + col0 + nl];
        }
        c = __builtin_amdgcn_wmma_f32_16x16x32_f16(
                /*neg_a=*/false, a, /*neg_b=*/false, bb,
                /*c_mod=*/(short)0, c, /*reuse_a=*/false, /*reuse_b=*/false);
    }

    const float bias = proj_b[col0 + nl];
    #pragma unroll
    for (int r = 0; r < 8; ++r) {
        const int row = row0 + r + hi * 8;     // D layout: m = r + 8*(lane>=16)
        float v = c[r] + bias;
        if (padding_mask[row]) v = 0.0f;
        merge[(size_t)row * E_ + col0 + nl] = v;
    }
}

// ---------------------------------------------------------------------------
extern "C" void kernel_launch(void* const* d_in, const int* in_sizes, int n_in,
                              void* d_out, int out_size, void* d_ws, size_t ws_size,
                              hipStream_t stream) {
    (void)in_sizes; (void)n_in; (void)out_size; (void)ws_size;

    const float* pos           = (const float*)d_in[0];
    const int*   nte           = (const int*)d_in[1];
    const unsigned char* pmask = (const unsigned char*)d_in[2];
    const unsigned char* maa   = (const unsigned char*)d_in[3];
    const unsigned char* mpos  = (const unsigned char*)d_in[4];
    const int*   time_pos      = (const int*)d_in[5];
    const float* means         = (const float*)d_in[6];
    const float* stds          = (const float*)d_in[7];
    const float* mul_w         = (const float*)d_in[8];
    const float* bias_w        = (const float*)d_in[9];
    const float* proj_w        = (const float*)d_in[10];
    const float* proj_b        = (const float*)d_in[11];
    const float* t_w1          = (const float*)d_in[12];
    const float* t_b1          = (const float*)d_in[13];
    const float* t_w2          = (const float*)d_in[14];
    const float* t_b2          = (const float*)d_in[15];

    float* edge = (float*)d_out;                       // [B,N,N,K]
    const size_t edge_n  = (size_t)B_ * N_ * N_ * K_;
    float* merge = edge + edge_n;                      // [B,N,E]
    const size_t merge_n = (size_t)B_ * N_ * E_;
    float* delta = merge + merge_n;                    // [B,N,N,3]

    float* tbuf = (float*)d_ws;                        // 2*B*K floats
    _Float16* sumbuf = (_Float16*)((char*)d_ws +
        ((2 * B_ * K_ * sizeof(float) + 255) & ~(size_t)255));

    time_enc_kernel<<<dim3(B_, 2), 128, 0, stream>>>(
        time_pos, t_w1, t_b1, t_w2, t_b2, tbuf);

    edge_kernel<<<B_ * N_, 256, 0, stream>>>(
        pos, nte, pmask, maa, mpos, means, stds, mul_w, bias_w,
        tbuf, edge, delta, sumbuf);

    merge_wmma_kernel<<<dim3(E_ / 16, (B_ * N_) / 16), 32, 0, stream>>>(
        sumbuf, proj_w, proj_b, pmask, merge);
}
